// HGN_56538949484990
// MI455X (gfx1250) — matrix operand order repaired and checked by reading, not compiled, not run on hardware
//
#include <hip/hip_runtime.h>
#include <hip/hip_bf16.h>

typedef __attribute__((ext_vector_type(16))) _Float16 v16h;
typedef __attribute__((ext_vector_type(8)))  float    v8f;

#define HEADS 4
#define HID   32
#define CH    128            // HEADS*HID == IN_CH == 128
#define NEGS  5

// ---------------- ordered-float helpers for atomic max on floats -----------
__device__ __forceinline__ unsigned f2ord(float f) {
    unsigned u = __float_as_uint(f);
    return (u & 0x80000000u) ? ~u : (u | 0x80000000u);
}
__device__ __forceinline__ float ord2f(unsigned u) {
    unsigned v = (u & 0x80000000u) ? (u & 0x7FFFFFFFu) : ~u;
    return __uint_as_float(v);
}

// ---------------- dual GEMM: feat = X@W, accum = X@Wres  (WMMA f16) --------
// One wave per (16-row tile) x (one full 128-col weight matrix).
// grid = (N/16, 2): y==0 -> W/feat, y==1 -> Wres/accum.
// Per wave: 4 K-steps x 8 column tiles = 32 v_wmma, A fragment reused 8x.
__global__ __launch_bounds__(32) void gemm_dual_wmma(
    const float* __restrict__ X, const float* __restrict__ W,
    const float* __restrict__ Wr, float* __restrict__ feat,
    float* __restrict__ accum, int N)
{
    const int rowTile = blockIdx.x;
    const int grp     = blockIdx.y;
    const float* Wsel = grp ? Wr : W;
    float*       Out  = grp ? accum : feat;
    const int lane = threadIdx.x;
    const int l15  = lane & 15;
    const int hi   = lane >> 4;
    const int row  = rowTile * 16 + l15;

    v8f c[8];
    #pragma unroll
    for (int t = 0; t < 8; ++t) c[t] = (v8f){};

    #pragma unroll
    for (int kb = 0; kb < 4; ++kb) {
        const int k0 = kb * 32;
        v16h a;
        // A 16x32 f16 layout (ISA 7.12.2): lanes 0-15 row M=l15 hold K={0..7,16..23},
        // lanes 16-31 hold K={8..15,24..31}
        #pragma unroll
        for (int i = 0; i < 8; ++i) {
            a[i]     = (_Float16)X[row * CH + k0 + hi * 8 + i];
            a[i + 8] = (_Float16)X[row * CH + k0 + 16 + hi * 8 + i];
        }
        #pragma unroll
        for (int t = 0; t < 8; ++t) {
            v16h b;
            // B 32x16 f16 layout: lane col N=l15; lanes 0-15 K={0..7,16..23}, 16-31 K={8..15,24..31}
            #pragma unroll
            for (int i = 0; i < 16; ++i)
                b[i] = (_Float16)Wsel[(k0 + hi * 8 + i + (i & 8)) * CH + t * 16 + l15];
            c[t] = __builtin_amdgcn_wmma_f32_16x16x32_f16(
                       false, a, false, b, (short)0, c[t], false, false);
        }
    }
    // C 16x16 f32 layout: VGPR r -> row M = r + 8*hi, col N = l15
    #pragma unroll
    for (int t = 0; t < 8; ++t) {
        #pragma unroll
        for (int r = 0; r < 8; ++r)
            Out[(rowTile * 16 + r + 8 * hi) * CH + t * 16 + l15] = c[t][r];
    }
}

// ---------------- per-node attention dots el/er ----------------------------
__global__ void node_attn_kernel(const float* __restrict__ feat,
                                 const float* __restrict__ al,
                                 const float* __restrict__ ar,
                                 float* __restrict__ el, float* __restrict__ er,
                                 int N)
{
    int idx = blockIdx.x * blockDim.x + threadIdx.x;
    if (idx >= N * HEADS) return;
    int n = idx >> 2, h = idx & 3;
    const float* f = feat + n * CH + h * HID;
    float sl = 0.f, sr = 0.f;
    #pragma unroll
    for (int d = 0; d < HID; ++d) {
        float v = f[d];
        sl += v * al[h * HID + d];
        sr += v * ar[h * HID + d];
    }
    el[idx] = sl;
    er[idx] = sr;
}

// ---------------- tiny edge-type term ee[t][h] -----------------------------
__global__ void ee_kernel(const float* __restrict__ eemb,
                          const float* __restrict__ We,
                          const float* __restrict__ ae,
                          float* __restrict__ ee)
{
    if (threadIdx.x >= 16) return;
    int t = threadIdx.x >> 2, h = threadIdx.x & 3;
    float s = 0.f;
    for (int d = 0; d < 32; ++d) {
        float efv = 0.f;
        for (int c = 0; c < 32; ++c)
            efv += eemb[t * 32 + c] * We[c * CH + h * 32 + d];
        s += efv * ae[h * 32 + d];
    }
    ee[t * 4 + h] = s;
}

// ---------------- init max (-inf ordered) / denom (0) ----------------------
__global__ void init_nodes_kernel(unsigned* __restrict__ maxb,
                                  float* __restrict__ denom, int n)
{
    int i = blockIdx.x * blockDim.x + threadIdx.x;
    if (i >= n) return;
    maxb[i]  = 0x007FFFFFu;   // f2ord(-inf)
    denom[i] = 0.f;
}

// ---------------- edge pass 1: logits + segment max ------------------------
__global__ void edge_logits_max_kernel(const int* __restrict__ src,
                                       const int* __restrict__ dst,
                                       const int* __restrict__ etype,
                                       const float* __restrict__ el,
                                       const float* __restrict__ er,
                                       const float* __restrict__ ee,
                                       float* __restrict__ edge_a,
                                       unsigned* __restrict__ maxb, int E)
{
    int e = blockIdx.x * blockDim.x + threadIdx.x;
    if (e >= E) return;
    int s = src[e], d = dst[e], t = etype[e];
    float4 elv = ((const float4*)el)[s];
    float4 erv = ((const float4*)er)[d];
    float4 eev = ((const float4*)ee)[t];
    float4 x;
    x.x = elv.x + erv.x + eev.x;
    x.y = elv.y + erv.y + eev.y;
    x.z = elv.z + erv.z + eev.z;
    x.w = elv.w + erv.w + eev.w;
    x.x = (x.x > 0.f) ? x.x : 0.2f * x.x;
    x.y = (x.y > 0.f) ? x.y : 0.2f * x.y;
    x.z = (x.z > 0.f) ? x.z : 0.2f * x.z;
    x.w = (x.w > 0.f) ? x.w : 0.2f * x.w;
    ((float4*)edge_a)[e] = x;
    atomicMax(&maxb[d * 4 + 0], f2ord(x.x));
    atomicMax(&maxb[d * 4 + 1], f2ord(x.y));
    atomicMax(&maxb[d * 4 + 2], f2ord(x.z));
    atomicMax(&maxb[d * 4 + 3], f2ord(x.w));
}

// ---------------- edge pass 2: exp + segment sum ---------------------------
__global__ void edge_exp_sum_kernel(const int* __restrict__ dst,
                                    float* __restrict__ edge_a,
                                    const unsigned* __restrict__ maxb,
                                    float* __restrict__ denom, int E)
{
    int e = blockIdx.x * blockDim.x + threadIdx.x;
    if (e >= E) return;
    int d = dst[e];
    float4 av = ((const float4*)edge_a)[e];
    uint4  mv = ((const uint4*)maxb)[d];
    av.x = __expf(av.x - ord2f(mv.x));
    av.y = __expf(av.y - ord2f(mv.y));
    av.z = __expf(av.z - ord2f(mv.z));
    av.w = __expf(av.w - ord2f(mv.w));
    ((float4*)edge_a)[e] = av;
    atomicAdd(&denom[d * 4 + 0], av.x);
    atomicAdd(&denom[d * 4 + 1], av.y);
    atomicAdd(&denom[d * 4 + 2], av.z);
    atomicAdd(&denom[d * 4 + 3], av.w);
}

// ---------------- edge pass 3: weighted scatter (L2-resident atomics) ------
__global__ void edge_scatter_kernel(const int* __restrict__ src,
                                    const int* __restrict__ dst,
                                    const float* __restrict__ feat,
                                    const float* __restrict__ edge_a,
                                    const float* __restrict__ denom,
                                    float* __restrict__ accum, int E)
{
    long long gid = (long long)blockIdx.x * blockDim.x + threadIdx.x;
    int e = (int)(gid >> 7);      // /128 channels
    int c = (int)(gid & 127);
    if (e >= E) return;
    int h = c >> 5;
    int s = src[e], d = dst[e];
    float alpha = edge_a[e * 4 + h] / (denom[d * 4 + h] + 1e-16f);
    atomicAdd(&accum[d * CH + c], feat[s * CH + c] * alpha);
}

// ---------------- finalize: x = (relu?) accum ------------------------------
__global__ void finalize_kernel(const float* __restrict__ accum,
                                float* __restrict__ out, int n, int do_relu)
{
    int i = blockIdx.x * blockDim.x + threadIdx.x;
    if (i >= n) return;
    float v = accum[i];
    out[i] = do_relu ? (v > 0.f ? v : 0.f) : v;
}

// ---------------- link-pred head: one wave per output row ------------------
__global__ __launch_bounds__(256) void head_kernel(
    const float* __restrict__ x, const int* __restrict__ n_id,
    const float* __restrict__ pW1, const float* __restrict__ pb1,
    const float* __restrict__ pW2, const float* __restrict__ pb2,
    float* __restrict__ out, int b)
{
    int wave = (int)((blockIdx.x * (long long)blockDim.x + threadIdx.x) >> 5);
    int lane = threadIdx.x & 31;
    int total = b * (1 + NEGS);
    if (wave >= total) return;
    int u, v, oidx;
    if (wave < b) { u = n_id[wave]; v = n_id[b + wave]; oidx = wave; }
    else {
        int j = wave - b;
        u = n_id[j / NEGS];            // repeat(out, NEG)
        v = n_id[2 * b + j];
        oidx = b + j;
    }
    // hidden[lane] = leaky( concat(x_u, x_v) . pW1[:,lane] + pb1[lane] )
    float hsum = pb1[lane];
    const float* xu = x + (long long)u * CH;
    const float* xv = x + (long long)v * CH;
    for (int k = 0; k < CH; ++k) hsum += xu[k] * pW1[k * 32 + lane];
    for (int k = 0; k < CH; ++k) hsum += xv[k] * pW1[(CH + k) * 32 + lane];
    hsum = (hsum > 0.f) ? hsum : 0.2f * hsum;
    float p = hsum * pW2[lane];
    #pragma unroll
    for (int off = 16; off > 0; off >>= 1) p += __shfl_down(p, off, 32);
    if (lane == 0) out[oidx] = p + pb2[0];
}

extern "C" void kernel_launch(void* const* d_in, const int* in_sizes, int n_in,
                              void* d_out, int out_size, void* d_ws, size_t ws_size,
                              hipStream_t stream)
{
    const int E = in_sizes[1];                 // 800000
    const int N = in_sizes[4] / CH;            // 100000
    const int b = in_sizes[2] / (2 + NEGS);    // 1024

    const int* edge_index = (const int*)d_in[0];
    const int* src   = edge_index;
    const int* dst   = edge_index + E;
    const int* etype = (const int*)d_in[1];
    const int* n_id  = (const int*)d_in[2];
    const float* id_emb = (const float*)d_in[4];
    const float* W0    = (const float*)d_in[5];
    const float* Wres0 = (const float*)d_in[6];
    const float* al0   = (const float*)d_in[7];
    const float* ar0   = (const float*)d_in[8];
    const float* ae0   = (const float*)d_in[9];
    const float* eemb0 = (const float*)d_in[10];
    const float* We0   = (const float*)d_in[11];
    const float* W1    = (const float*)d_in[12];
    const float* Wres1 = (const float*)d_in[13];
    const float* al1   = (const float*)d_in[14];
    const float* ar1   = (const float*)d_in[15];
    const float* ae1   = (const float*)d_in[16];
    const float* eemb1 = (const float*)d_in[17];
    const float* We1   = (const float*)d_in[18];
    const float* pW1   = (const float*)d_in[19];
    const float* pb1   = (const float*)d_in[20];
    const float* pW2   = (const float*)d_in[21];
    const float* pb2   = (const float*)d_in[22];

    // workspace carve-up (~173 MB; node working set is L2-resident on MI455X)
    const long long NF = (long long)N * CH;    // 12.8M floats
    const long long N4 = (long long)N * HEADS;
    const long long E4 = (long long)E * HEADS;
    float* ws     = (float*)d_ws;
    float* feat   = ws;
    float* accum  = feat  + NF;
    float* xbuf   = accum + NF;
    float* el     = xbuf  + NF;
    float* er     = el + N4;
    unsigned* maxb = (unsigned*)(er + N4);
    float* denom  = (float*)maxb + N4;
    float* edge_a = denom + N4;
    float* ee0    = edge_a + E4;
    float* ee1    = ee0 + 16;

    ee_kernel<<<1, 16, 0, stream>>>(eemb0, We0, ae0, ee0);
    ee_kernel<<<1, 16, 0, stream>>>(eemb1, We1, ae1, ee1);

    const int T = 256;
    auto run_layer = [&](const float* Xin, const float* W, const float* Wres,
                         const float* al, const float* ar, const float* ee,
                         float* outX, int do_relu) {
        gemm_dual_wmma<<<dim3(N / 16, 2), 32, 0, stream>>>(Xin, W, Wres,
                                                           feat, accum, N);
        node_attn_kernel<<<(int)((N4 + T - 1) / T), T, 0, stream>>>(feat, al, ar,
                                                                    el, er, N);
        init_nodes_kernel<<<(int)((N4 + T - 1) / T), T, 0, stream>>>(maxb, denom,
                                                                     (int)N4);
        edge_logits_max_kernel<<<(E + T - 1) / T, T, 0, stream>>>(
            src, dst, etype, el, er, ee, edge_a, maxb, E);
        edge_exp_sum_kernel<<<(E + T - 1) / T, T, 0, stream>>>(
            dst, edge_a, maxb, denom, E);
        long long work = (long long)E * CH;
        edge_scatter_kernel<<<(int)((work + T - 1) / T), T, 0, stream>>>(
            src, dst, feat, edge_a, denom, accum, E);
        finalize_kernel<<<(int)((NF + T - 1) / T), T, 0, stream>>>(
            accum, outX, (int)NF, do_relu);
    };

    run_layer(id_emb, W0, Wres0, al0, ar0, ee0, xbuf, 1);  // layer 0 + ReLU
    run_layer(xbuf,   W1, Wres1, al1, ar1, ee1, xbuf, 0);  // layer 1

    int total = b * (1 + NEGS);                            // 6144 rows
    head_kernel<<<(total * 32 + T - 1) / T, T, 0, stream>>>(
        xbuf, n_id, pW1, pb1, pW2, pb2, (float*)d_out, b);
}
